// XEyTransformerLayer_74388833567211
// MI455X (gfx1250) — compile-verified, matrix operand order checked by reading, not run on hardware
//
#include <hip/hip_runtime.h>
#include <hip/hip_bf16.h>

// ---------------------------------------------------------------------------
// XEyTransformerLayer forward for MI455X (gfx1250, wave32, WMMA).
// All K%32==0 contractions run on v_wmma_f32_16x16x32_f16 (f16 A/B, f32 acc).
// GEMM: 4-wave blocks, 64x64 macro-tile, LDS-staged f16 B, b128 A loads.
// ---------------------------------------------------------------------------

typedef __attribute__((ext_vector_type(16))) _Float16 v16h;
typedef __attribute__((ext_vector_type(8)))  _Float16 h8;
typedef __attribute__((ext_vector_type(8)))  float    v8f;

constexpr int CB  = 8;     // batch
constexpr int CN  = 128;   // nodes / graph
constexpr int CDX = 256;
constexpr int CDE = 64;
constexpr int CDY = 64;
constexpr int CNH = 8;
constexpr int CFFX = 2048;
constexpr int CFFE = 128;
constexpr int CFFY = 2048;
constexpr int NODES  = CB * CN;            // 1024
constexpr int EDGESI = CB * CN * CN;       // 131072
constexpr float BIGC  = 100000.0f;
constexpr float EPSLN = 1e-5f;

// ---- fragment loaders (wave32 16x16x32 f16 WMMA layout) -------------------
// Lane (half = lane>>4, mr/nc = lane&15) holds K = {8h+0..7} and {16+8h+0..7}.

// A fragment from a contiguous fp32 row: 4x global_load_b128 + cvt.
__device__ __forceinline__ v16h load_a_row(const float* __restrict__ ap, int half)
{
    const float4* p0 = (const float4*)(ap + 8 * half);
    const float4* p1 = (const float4*)(ap + 16 + 8 * half);
    float4 x0 = p0[0], x1 = p0[1], x2 = p1[0], x3 = p1[1];
    v16h f;
    f[0]=(_Float16)x0.x; f[1]=(_Float16)x0.y; f[2]=(_Float16)x0.z; f[3]=(_Float16)x0.w;
    f[4]=(_Float16)x1.x; f[5]=(_Float16)x1.y; f[6]=(_Float16)x1.z; f[7]=(_Float16)x1.w;
    f[8]=(_Float16)x2.x; f[9]=(_Float16)x2.y; f[10]=(_Float16)x2.z; f[11]=(_Float16)x2.w;
    f[12]=(_Float16)x3.x; f[13]=(_Float16)x3.y; f[14]=(_Float16)x3.z; f[15]=(_Float16)x3.w;
    return f;
}

// B fragment from a strided fp32 column: batched loads, then converts.
__device__ __forceinline__ v16h load_b_col(const float* __restrict__ wp, int ldw, int half)
{
    float r[16];
#pragma unroll
    for (int i = 0; i < 8; ++i) r[i]     = wp[(size_t)(8 * half + i) * ldw];
#pragma unroll
    for (int i = 0; i < 8; ++i) r[8 + i] = wp[(size_t)(16 + 8 * half + i) * ldw];
    v16h f;
#pragma unroll
    for (int i = 0; i < 16; ++i) f[i] = (_Float16)r[i];
    return f;
}

// ---------------------------------------------------------------------------
// Generic WMMA GEMM: C[M,N] = epi(A[M,K] @ W[K,N] + bias)
// Block = 128 threads (4 waves); macro-tile 64(M) x 64(N); K stepped by 32.
// Requires M%64==0, N%64==0, K%32==0 (true for every call site).
// B tile staged to LDS as f16 [n][k]; lanes read it with 2x ds_load_b128.
// ---------------------------------------------------------------------------
__global__ __launch_bounds__(128)
void k_gemm(const float* __restrict__ A, int lda,
            const float* __restrict__ Wt, int ldw,
            const float* __restrict__ bias,
            float* __restrict__ C, int ldc,
            int M, int Nn, int K,
            const unsigned char* __restrict__ rowmask,
            int relu)
{
    __shared__ __align__(16) _Float16 LT[64 * 32];   // [n][k], 4 KB
    const int tid  = threadIdx.x;
    const int wave = tid >> 5, lane = tid & 31;
    const int half = lane >> 4, mr = lane & 15, nc = lane & 15;
    const int m0 = blockIdx.y * 64 + wave * 16;
    const int n0 = blockIdx.x * 64;
    const float* arow = A + (size_t)(m0 + mr) * lda;
    const int sr = tid >> 2;            // staging row (k) 0..31
    const int sc = (tid & 3) * 16;      // staging col chunk
    v8f acc[4] = {};
    for (int k0 = 0; k0 < K; k0 += 32) {
        // ---- stage B tile (coalesced b128 reads, f16 transpose into LDS) --
        const float* wp = Wt + (size_t)(k0 + sr) * ldw + n0 + sc;
        float4 w0 = ((const float4*)wp)[0];
        float4 w1 = ((const float4*)wp)[1];
        float4 w2 = ((const float4*)wp)[2];
        float4 w3 = ((const float4*)wp)[3];
        __syncthreads();                 // previous iter readers done
        _Float16* lt = &LT[sr];
        lt[(sc+ 0)*32]=(_Float16)w0.x; lt[(sc+ 1)*32]=(_Float16)w0.y;
        lt[(sc+ 2)*32]=(_Float16)w0.z; lt[(sc+ 3)*32]=(_Float16)w0.w;
        lt[(sc+ 4)*32]=(_Float16)w1.x; lt[(sc+ 5)*32]=(_Float16)w1.y;
        lt[(sc+ 6)*32]=(_Float16)w1.z; lt[(sc+ 7)*32]=(_Float16)w1.w;
        lt[(sc+ 8)*32]=(_Float16)w2.x; lt[(sc+ 9)*32]=(_Float16)w2.y;
        lt[(sc+10)*32]=(_Float16)w2.z; lt[(sc+11)*32]=(_Float16)w2.w;
        lt[(sc+12)*32]=(_Float16)w3.x; lt[(sc+13)*32]=(_Float16)w3.y;
        lt[(sc+14)*32]=(_Float16)w3.z; lt[(sc+15)*32]=(_Float16)w3.w;
        __syncthreads();
        // ---- A fragment (vector loads), shared across 4 N-subtiles -------
        v16h av = load_a_row(arow + k0, half);
        if (k0 + 64 < K)                 // uniform branch: cheap scalar skip
            __builtin_prefetch(arow + k0 + 64, 0, 3);
#pragma unroll
        for (int nt = 0; nt < 4; ++nt) {
            const h8* bp0 = (const h8*)&LT[(nt * 16 + nc) * 32 + 8 * half];
            const h8* bp1 = (const h8*)&LT[(nt * 16 + nc) * 32 + 16 + 8 * half];
            v16h bv = __builtin_shufflevector(bp0[0], bp1[0],
                        0,1,2,3,4,5,6,7,8,9,10,11,12,13,14,15);
            acc[nt] = __builtin_amdgcn_wmma_f32_16x16x32_f16(false, av, false, bv,
                                                             (short)0, acc[nt], false, false);
        }
    }
#pragma unroll
    for (int nt = 0; nt < 4; ++nt) {
        int col = n0 + nt * 16 + nc;
        float bb = bias ? bias[col] : 0.f;
#pragma unroll
        for (int r = 0; r < 8; ++r) {
            int row = m0 + r + 8 * half;
            float v = acc[nt][r] + bb;
            if (relu) v = fmaxf(v, 0.f);
            if (rowmask) v *= rowmask[row] ? 1.f : 0.f;
            C[(size_t)row * ldc + col] = v;
        }
    }
}

// ---------------------------------------------------------------------------
// Small-N WMMA GEMM (N <= 16): one wave per 16-row tile; loads are guard-free
// (column index clamped, uniform flow); only the final store is guarded.
// ---------------------------------------------------------------------------
__global__ __launch_bounds__(32)
void k_gemm_s(const float* __restrict__ A, int lda,
              const float* __restrict__ Wt, int ldw,
              const float* __restrict__ bias,
              float* __restrict__ C, int ldc,
              int M, int Nn, int K)
{
    const int m0 = blockIdx.x * 16;
    const int lane = threadIdx.x & 31;
    const int half = lane >> 4, mr = lane & 15, nc = lane & 15;
    const int ccol = nc < Nn ? nc : Nn - 1;
    const float* arow = A + (size_t)(m0 + mr) * lda;
    v8f acc = {};
    for (int k0 = 0; k0 < K; k0 += 32) {
        v16h av = load_a_row(arow + k0, half);
        v16h bv = load_b_col(Wt + (size_t)k0 * ldw + ccol, ldw, half);
        acc = __builtin_amdgcn_wmma_f32_16x16x32_f16(false, av, false, bv,
                                                     (short)0, acc, false, false);
    }
    if (nc >= Nn) return;
    float bb = bias ? bias[nc] : 0.f;
#pragma unroll
    for (int r = 0; r < 8; ++r)
        C[(size_t)(m0 + r + 8 * half) * ldc + nc] = acc[r] + bb;
}

// ---------------------------------------------------------------------------
// Attention logits into (b,h,i,j) layout:
//   a0[b,h,i,j] = sum_d Q[b,i,d]*K[b,j,d]*Wa[d,h] / 8
// One wave per 16-j tile of fixed (b,i); A built on the fly from b128 loads.
// ---------------------------------------------------------------------------
__global__ __launch_bounds__(32)
void k_att_logits(const float* __restrict__ Q, const float* __restrict__ Kx,
                  const float* __restrict__ Wa, float* __restrict__ a0)
{
    int tile = blockIdx.x;                 // EDGES/16 = 8192
    int j0 = (tile & 7) * 16;
    int bi = tile >> 3;                    // b*128+i
    int b  = bi >> 7, i = bi & 127;
    const int lane = threadIdx.x & 31;
    const int half = lane >> 4, mr = lane & 15, nc = lane & 15;
    const int hh = nc < CNH ? nc : CNH - 1;
    const float* qrow = Q + (size_t)bi * CDX;
    const float* krow = Kx + (size_t)(b * CN + j0 + mr) * CDX;
    v8f acc = {};
    for (int k0 = 0; k0 < CDX; k0 += 32) {
        const float4* qp0 = (const float4*)(qrow + k0 + 8 * half);
        const float4* qp1 = (const float4*)(qrow + k0 + 16 + 8 * half);
        const float4* kp0 = (const float4*)(krow + k0 + 8 * half);
        const float4* kp1 = (const float4*)(krow + k0 + 16 + 8 * half);
        float4 q0 = qp0[0], q1 = qp0[1], q2 = qp1[0], q3 = qp1[1];
        float4 c0 = kp0[0], c1 = kp0[1], c2 = kp1[0], c3 = kp1[1];
        v16h av;
        av[0]=(_Float16)(q0.x*c0.x*0.125f); av[1]=(_Float16)(q0.y*c0.y*0.125f);
        av[2]=(_Float16)(q0.z*c0.z*0.125f); av[3]=(_Float16)(q0.w*c0.w*0.125f);
        av[4]=(_Float16)(q1.x*c1.x*0.125f); av[5]=(_Float16)(q1.y*c1.y*0.125f);
        av[6]=(_Float16)(q1.z*c1.z*0.125f); av[7]=(_Float16)(q1.w*c1.w*0.125f);
        av[8]=(_Float16)(q2.x*c2.x*0.125f); av[9]=(_Float16)(q2.y*c2.y*0.125f);
        av[10]=(_Float16)(q2.z*c2.z*0.125f); av[11]=(_Float16)(q2.w*c2.w*0.125f);
        av[12]=(_Float16)(q3.x*c3.x*0.125f); av[13]=(_Float16)(q3.y*c3.y*0.125f);
        av[14]=(_Float16)(q3.z*c3.z*0.125f); av[15]=(_Float16)(q3.w*c3.w*0.125f);
        v16h bv = load_b_col(Wa + (size_t)k0 * CNH + hh, CNH, half);
        acc = __builtin_amdgcn_wmma_f32_16x16x32_f16(false, av, false, bv,
                                                     (short)0, acc, false, false);
    }
    if (nc >= CNH) return;
#pragma unroll
    for (int r = 0; r < 8; ++r) {
        int j = j0 + r + 8 * half;
        a0[((size_t)(b * CNH + nc) * CN + i) * CN + j] = acc[r];
    }
}

// ---------------------------------------------------------------------------
// wV[b,i,h*256+d] = sum_j alpha[b,h,i,j] * V[b,j,d]   (batched WMMA, K=128)
// alpha rows are contiguous in j -> b128 A loads.
// ---------------------------------------------------------------------------
__global__ __launch_bounds__(32)
void k_wV(const float* __restrict__ alpha, const float* __restrict__ V,
          float* __restrict__ wV)
{
    int b = blockIdx.z >> 3, h = blockIdx.z & 7;
    int n0 = blockIdx.x * 16, i0 = blockIdx.y * 16;
    const int lane = threadIdx.x & 31;
    const int half = lane >> 4, mr = lane & 15, nc = lane & 15;
    const float* arow = alpha + ((size_t)(b * CNH + h) * CN + i0 + mr) * CN;
    v8f acc = {};
    for (int k0 = 0; k0 < CN; k0 += 32) {
        v16h av = load_a_row(arow + k0, half);
        v16h bv = load_b_col(V + (size_t)(b * CN + k0) * CDX + n0 + nc, CDX, half);
        acc = __builtin_amdgcn_wmma_f32_16x16x32_f16(false, av, false, bv,
                                                     (short)0, acc, false, false);
    }
#pragma unroll
    for (int r = 0; r < 8; ++r) {
        int i = i0 + r + 8 * half;
        wV[((size_t)b * CN + i) * (CNH * CDX) + h * CDX + n0 + nc] = acc[r];
    }
}

// ---------------------------------------------------------------------------
// Fused edge FFN: out = linE2(relu(linE1(E1)))  (64 -> 128 -> 64)
// one wave per 16-edge tile; hidden tile in LDS as f16; stage-2 A fragments
// read back as 2x ds_load_b128.
// ---------------------------------------------------------------------------
__global__ __launch_bounds__(32)
void k_effn(const float* __restrict__ E1,
            const float* __restrict__ W1, const float* __restrict__ B1,
            const float* __restrict__ W2, const float* __restrict__ B2,
            float* __restrict__ out)
{
    __shared__ __align__(16) _Float16 hl[16 * CFFE];
    size_t e0 = (size_t)blockIdx.x * 16;
    const int lane = threadIdx.x & 31;
    const int half = lane >> 4, mr = lane & 15, nc = lane & 15;
    const float* arow = E1 + (e0 + mr) * CDE;
    // stage 1: H = relu(E1 @ W1 + B1), K=64, N=128
    for (int nt = 0; nt < CFFE / 16; ++nt) {
        int n0 = nt * 16;
        v8f acc = {};
        for (int k0 = 0; k0 < CDE; k0 += 32) {
            v16h av = load_a_row(arow + k0, half);
            v16h bv = load_b_col(W1 + (size_t)k0 * CFFE + n0 + nc, CFFE, half);
            acc = __builtin_amdgcn_wmma_f32_16x16x32_f16(false, av, false, bv,
                                                         (short)0, acc, false, false);
        }
        float bb = B1[n0 + nc];
#pragma unroll
        for (int r = 0; r < 8; ++r)
            hl[(r + 8 * half) * CFFE + n0 + nc] = (_Float16)fmaxf(acc[r] + bb, 0.f);
    }
    __syncthreads();
    // stage 2: out = H @ W2 + B2, K=128, N=64 (A from LDS via ds_load_b128)
    for (int nt = 0; nt < CDE / 16; ++nt) {
        int n0 = nt * 16;
        v8f acc = {};
        for (int k0 = 0; k0 < CFFE; k0 += 32) {
            const h8* hp0 = (const h8*)&hl[mr * CFFE + k0 + 8 * half];
            const h8* hp1 = (const h8*)&hl[mr * CFFE + k0 + 16 + 8 * half];
            v16h av = __builtin_shufflevector(hp0[0], hp1[0],
                        0,1,2,3,4,5,6,7,8,9,10,11,12,13,14,15);
            v16h bv = load_b_col(W2 + (size_t)k0 * CDE + n0 + nc, CDE, half);
            acc = __builtin_amdgcn_wmma_f32_16x16x32_f16(false, av, false, bv,
                                                         (short)0, acc, false, false);
        }
        float bb = B2[n0 + nc];
#pragma unroll
        for (int r = 0; r < 8; ++r)
            out[(e0 + r + 8 * half) * CDE + n0 + nc] = acc[r] + bb;
    }
}

// ------------------------------ small kernels ------------------------------

__global__ void k_nval(const unsigned char* __restrict__ mask, float* __restrict__ nval)
{
    __shared__ float s[CN];
    int b = blockIdx.x, t = threadIdx.x;
    s[t] = mask[b * CN + t] ? 1.f : 0.f;
    __syncthreads();
    for (int o = CN / 2; o > 0; o >>= 1) { if (t < o) s[t] += s[t + o]; __syncthreads(); }
    if (t == 0) nval[b] = s[0];
}

__global__ void k_edgemask(const unsigned char* __restrict__ mask, unsigned char* __restrict__ em)
{
    int e = blockIdx.x * 256 + threadIdx.x;
    if (e >= EDGESI) return;
    int b = e >> 14, i = (e >> 7) & 127, j = e & 127;
    em[e] = (mask[b * CN + i] && mask[b * CN + j]) ? 1 : 0;
}

__global__ void k_prep_node(const float* __restrict__ pos, const unsigned char* __restrict__ mask,
                            const float* __restrict__ w1, const float* __restrict__ b1,
                            const float* __restrict__ w2, const float* __restrict__ b2,
                            float* __restrict__ posm, float* __restrict__ normpos,
                            float* __restrict__ np1, float* __restrict__ np2)
{
    int node = blockIdx.x, t = threadIdx.x;
    float m = mask[node] ? 1.f : 0.f;
    float px = pos[node * 3 + 0] * m, py = pos[node * 3 + 1] * m, pz = pos[node * 3 + 2] * m;
    float s = px * px + py * py + pz * pz;
    float nrm = s > 0.f ? sqrtf(s) : 0.f;
    if (t == 0) { posm[node * 3 + 0] = px; posm[node * 3 + 1] = py; posm[node * 3 + 2] = pz; normpos[node] = nrm; }
    np1[node * CDE + t] = w1[t] * nrm + b1[t];
    np2[node * CDE + t] = w2[t] * nrm + b2[t];
}

__global__ void k_ylin(const float* __restrict__ y,
                       const float* __restrict__ wem, const float* __restrict__ bem,
                       const float* __restrict__ wea, const float* __restrict__ bea,
                       const float* __restrict__ wxm, const float* __restrict__ bxm,
                       const float* __restrict__ wxa, const float* __restrict__ bxa,
                       float* __restrict__ ye_mul, float* __restrict__ ye_add,
                       float* __restrict__ yx_mul, float* __restrict__ yx_add)
{
    int b = blockIdx.x, t = threadIdx.x;
    const float* yb = y + b * CDY;
    if (t < CDE) {
        float s1 = bem[t], s2 = bea[t];
        for (int k = 0; k < CDY; ++k) { float v = yb[k]; s1 += v * wem[k * CDE + t]; s2 += v * wea[k * CDE + t]; }
        ye_mul[b * CDE + t] = s1; ye_add[b * CDE + t] = s2;
    }
    float s3 = bxm[t], s4 = bxa[t];
    for (int k = 0; k < CDY; ++k) { float v = yb[k]; s3 += v * wxm[k * CDX + t]; s4 += v * wxa[k * CDX + t]; }
    yx_mul[b * CDX + t] = s3; yx_add[b * CDX + t] = s4;
}

__global__ void k_edge_geom(const float* __restrict__ posm, const float* __restrict__ normpos,
                            const float* __restrict__ np1, const float* __restrict__ np2,
                            const unsigned char* __restrict__ mask,
                            const float* __restrict__ wd, const float* __restrict__ bd,
                            float* __restrict__ dist1)
{
    int e = blockIdx.x, t = threadIdx.x;
    int b = e >> 14, i = (e >> 7) & 127, j = e & 127;
    int ni = b * CN + i, nj = b * CN + j;
    float pix = posm[ni * 3], piy = posm[ni * 3 + 1], piz = posm[ni * 3 + 2];
    float pjx = posm[nj * 3], pjy = posm[nj * 3 + 1], pjz = posm[nj * 3 + 2];
    float dx = pix - pjx, dy = piy - pjy, dz = piz - pjz;
    float d2 = dx * dx + dy * dy + dz * dz;
    float pdist = d2 > 0.f ? sqrtf(d2) : 0.f;
    float cosv = (pix * pjx + piy * pjy + piz * pjz) /
                 ((normpos[ni] + 1e-7f) * (normpos[nj] + 1e-7f));
    float eem = (mask[ni] && mask[nj]) ? 1.f : 0.f;
    float v = wd[t] * pdist + wd[CDE + t] * cosv + bd[t] + np1[ni * CDE + t] + np2[nj * CDE + t];
    dist1[(size_t)e * CDE + t] = fmaxf(v, 0.f) * eem;
}

__global__ void k_edge_combine(const float* __restrict__ Y0, const float* __restrict__ DA,
                               const float* __restrict__ DM,
                               const float* __restrict__ xe1, const float* __restrict__ xe2,
                               const float* __restrict__ ye_mul, const float* __restrict__ ye_add,
                               const unsigned char* __restrict__ mask,
                               float* __restrict__ Y, float* __restrict__ Econd)
{
    size_t idx = (size_t)blockIdx.x * 256 + threadIdx.x;
    if (idx >= (size_t)EDGESI * CDE) return;
    int c = idx & 63; size_t e = idx >> 6;
    int b = (int)(e >> 14), i = (int)((e >> 7) & 127), j = (int)(e & 127);
    float eem = (mask[b * CN + i] && mask[b * CN + j]) ? 1.f : 0.f;
    float y0 = Y0[idx] * xe1[(b * CN + j) * CDE + c] * xe2[(b * CN + i) * CDE + c] * eem;
    float yv = (y0 + DA[idx] + y0 * DM[idx]) * eem;
    Y[idx] = yv;
    Econd[idx] = (yv + ye_add[b * CDE + c] + yv * ye_mul[b * CDE + c]) * eem;
}

// softmax over j; a0/alpha in (b,h,i,j) layout, ea/pa in (b,i,j,h) layout.
__global__ void k_att_softmax(const float* __restrict__ a0, const float* __restrict__ ea,
                              const float* __restrict__ pa,
                              const unsigned char* __restrict__ mask,
                              float* __restrict__ alpha)
{
    int id = blockIdx.x;                 // (b*8+h)*128 + i
    int bh = id >> 7, i = id & 127;
    int b = bh >> 3, h = bh & 7;
    int lane = threadIdx.x;
    const float* arow = a0 + ((size_t)bh * CN + i) * CN;
    float* orow = alpha + ((size_t)bh * CN + i) * CN;
    float mi = mask[b * CN + i] ? 1.f : 0.f;
    float l[4]; float mx = -3.0e38f;
#pragma unroll
    for (int q = 0; q < 4; ++q) {
        int j = q * 32 + lane;
        size_t eidx = ((size_t)(b * CN + i) * CN + j) * CNH + h;
        float mj = mask[b * CN + j] ? 1.f : 0.f;
        float v = arow[j] * (1.f + ea[eidx]) * (1.f + pa[eidx]) * mi * mj;
        v = (mj != 0.f) ? v : -3.0e38f;
        l[q] = v; mx = fmaxf(mx, v);
    }
#pragma unroll
    for (int o = 16; o > 0; o >>= 1) mx = fmaxf(mx, __shfl_xor(mx, o, 32));
    float sum = 0.f;
#pragma unroll
    for (int q = 0; q < 4; ++q) { float ev = expf(l[q] - mx); l[q] = ev; sum += ev; }
#pragma unroll
    for (int o = 16; o > 0; o >>= 1) sum += __shfl_xor(sum, o, 32);
    float inv = 1.f / sum;
#pragma unroll
    for (int q = 0; q < 4; ++q) { int j = q * 32 + lane; orow[j] = l[q] * inv; }
}

__global__ void k_etox(const float* __restrict__ S, const unsigned char* __restrict__ mask,
                       const float* __restrict__ nval, float* __restrict__ out)
{   // per (b,i): mean/min/max/std over masked j -> (NODES,256)
    int ni = blockIdx.x, c = threadIdx.x, b = ni >> 7;
    float sum = 0.f, sq = 0.f, mn = 1e30f, mx = -1e30f;
    for (int j = 0; j < CN; ++j) {
        float v = S[((size_t)ni * CN + j) * CDE + c];
        float mj = mask[b * CN + j] ? 1.f : 0.f;
        sum += v; sq += v * v;
        mn = fminf(mn, (mj != 0.f) ? v : v + BIGC);
        mx = fmaxf(mx, (mj != 0.f) ? v : v - BIGC);
    }
    float nv = nval[b], m = sum / nv;
    float stdv = sq / nv - m * m;
    out[(size_t)ni * 256 + c]        = m;
    out[(size_t)ni * 256 + 64 + c]   = mn;
    out[(size_t)ni * 256 + 128 + c]  = mx;
    out[(size_t)ni * 256 + 192 + c]  = stdv;
}

__global__ void k_etoy1(const float* __restrict__ S, const unsigned char* __restrict__ mask,
                        float* __restrict__ st)
{
    int ni = blockIdx.x, c = threadIdx.x, b = ni >> 7;
    float mi = mask[ni] ? 1.f : 0.f;
    float sum = 0.f, sq = 0.f, mn = 1e30f, mx = -1e30f;
    for (int j = 0; j < CN; ++j) {
        float v = S[((size_t)ni * CN + j) * CDE + c];
        bool pm = (mi != 0.f) && mask[b * CN + j];
        sum += v; sq += v * v;
        mn = fminf(mn, pm ? v : v + BIGC);
        mx = fmaxf(mx, pm ? v : v - BIGC);
    }
    st[0 * (NODES * CDE) + ni * CDE + c] = sum;
    st[1 * (NODES * CDE) + ni * CDE + c] = sq;
    st[2 * (NODES * CDE) + ni * CDE + c] = mn;
    st[3 * (NODES * CDE) + ni * CDE + c] = mx;
}

__global__ void k_etoy2(const float* __restrict__ st, const float* __restrict__ nval,
                        float* __restrict__ out)
{
    int b = blockIdx.x, c = threadIdx.x;
    float sum = 0.f, sq = 0.f, mn = 1e30f, mx = -1e30f;
    for (int i = 0; i < CN; ++i) {
        int ni = b * CN + i;
        sum += st[ni * CDE + c];
        sq  += st[(NODES * CDE) + ni * CDE + c];
        mn = fminf(mn, st[2 * (NODES * CDE) + ni * CDE + c]);
        mx = fmaxf(mx, st[3 * (NODES * CDE) + ni * CDE + c]);
    }
    float nv = nval[b], nv2 = nv * nv, m = sum / nv2;
    float stdv = sq / nv2 - m * m;
    out[b * 256 + c] = m; out[b * 256 + 64 + c] = mn;
    out[b * 256 + 128 + c] = mx; out[b * 256 + 192 + c] = stdv;
}

__global__ void k_xtoy(const float* __restrict__ Xf, const unsigned char* __restrict__ mask,
                       const float* __restrict__ nval, float* __restrict__ out)
{
    int b = blockIdx.x, c = threadIdx.x;
    float sum = 0.f, sm = 0.f, sq = 0.f, mn = 1e30f, mx = -1e30f;
    for (int i = 0; i < CN; ++i) {
        float v = Xf[((size_t)b * CN + i) * CDX + c];
        float mi = mask[b * CN + i] ? 1.f : 0.f;
        sum += v; sm += v * mi; sq += v * v * mi;
        mn = fminf(mn, (mi != 0.f) ? v : v + BIGC);
        mx = fmaxf(mx, (mi != 0.f) ? v : v - BIGC);
    }
    float nv = nval[b], m = sum / nv;
    float stdv = (sq - 2.f * m * sm + m * m * nv) / nv;
    out[b * 1024 + c] = m; out[b * 1024 + 256 + c] = mn;
    out[b * 1024 + 512 + c] = mx; out[b * 1024 + 768 + c] = stdv;
}

__global__ void k_newX(const float* __restrict__ wVo, const float* __restrict__ exm,
                       const float* __restrict__ pxm, const float* __restrict__ yx_mul,
                       const float* __restrict__ yx_add, float* __restrict__ newX)
{
    int ni = blockIdx.x, c = threadIdx.x, b = ni >> 7;
    float v = wVo[(size_t)ni * CDX + c];
    v = v + exm[(size_t)ni * CDX + c] * v;
    v = v + pxm[(size_t)ni * CDX + c] * v;
    newX[(size_t)ni * CDX + c] = v * (yx_mul[b * CDX + c] + 1.f) + yx_add[b * CDX + c];
}

__global__ __launch_bounds__(32)
void k_ln_node(const float* __restrict__ x1, const float* __restrict__ x2,
               const float* __restrict__ g, const float* __restrict__ be,
               float* __restrict__ out)
{
    size_t nd = blockIdx.x; int lane = threadIdx.x;
    float v[8]; float s = 0.f;
#pragma unroll
    for (int q = 0; q < 8; ++q) { size_t idx = nd * CDX + q * 32 + lane; v[q] = x1[idx] + x2[idx]; s += v[q]; }
#pragma unroll
    for (int o = 16; o > 0; o >>= 1) s += __shfl_xor(s, o, 32);
    float mean = s * (1.f / 256.f), q2 = 0.f;
#pragma unroll
    for (int q = 0; q < 8; ++q) { v[q] -= mean; q2 += v[q] * v[q]; }
#pragma unroll
    for (int o = 16; o > 0; o >>= 1) q2 += __shfl_xor(q2, o, 32);
    float inv = rsqrtf(q2 * (1.f / 256.f) + EPSLN);
#pragma unroll
    for (int q = 0; q < 8; ++q) { int c = q * 32 + lane; out[nd * CDX + c] = v[q] * inv * g[c] + be[c]; }
}

__global__ __launch_bounds__(32)
void k_ln_edge(const float* __restrict__ x1, const float* __restrict__ x2,
               const float* __restrict__ g, const float* __restrict__ be,
               float* __restrict__ out)
{
    size_t e = blockIdx.x; int lane = threadIdx.x;
    float v0 = x1[e * CDE + lane]      + x2[e * CDE + lane];
    float v1 = x1[e * CDE + 32 + lane] + x2[e * CDE + 32 + lane];
    float s = v0 + v1;
#pragma unroll
    for (int o = 16; o > 0; o >>= 1) s += __shfl_xor(s, o, 32);
    float mean = s * (1.f / 64.f);
    float d0 = v0 - mean, d1 = v1 - mean;
    float q = d0 * d0 + d1 * d1;
#pragma unroll
    for (int o = 16; o > 0; o >>= 1) q += __shfl_xor(q, o, 32);
    float inv = rsqrtf(q * (1.f / 64.f) + EPSLN);
    out[e * CDE + lane]      = d0 * inv * g[lane] + be[lane];
    out[e * CDE + 32 + lane] = d1 * inv * g[32 + lane] + be[32 + lane];
}

__global__ void k_sym(const float* __restrict__ Ef0, float* __restrict__ EfOut)
{
    size_t e = blockIdx.x; int c = threadIdx.x;
    int b = (int)(e >> 14), i = (int)((e >> 7) & 127), j = (int)(e & 127);
    size_t et = ((size_t)b * CN + j) * CN + i;
    EfOut[e * CDE + c] = 0.5f * (Ef0[e * CDE + c] + Ef0[et * CDE + c]);
}

// ------------------------------ y path ------------------------------------

__global__ void k_ycombine(const float* __restrict__ y, const float* __restrict__ xstats,
                           const float* __restrict__ eY, const float* __restrict__ eD,
                           const float* __restrict__ wyy, const float* __restrict__ byy,
                           const float* __restrict__ wxy, const float* __restrict__ bxy,
                           const float* __restrict__ wey, const float* __restrict__ bey,
                           const float* __restrict__ wdy, const float* __restrict__ bdy,
                           float* __restrict__ new_y)
{
    int b = blockIdx.x, t = threadIdx.x;
    float s = byy[t] + bxy[t] + bey[t] + bdy[t];
    for (int k = 0; k < CDY; ++k)  s += y[b * CDY + k] * wyy[k * CDY + t];
    for (int k = 0; k < 1024; ++k) s += xstats[b * 1024 + k] * wxy[k * CDY + t];
    for (int k = 0; k < 256; ++k)  s += eY[b * 256 + k] * wey[k * CDY + t];
    for (int k = 0; k < 256; ++k)  s += eD[b * 256 + k] * wdy[k * CDY + t];
    new_y[b * CDY + t] = s;
}

__global__ void k_yout(const float* __restrict__ y, const float* __restrict__ new_y,
                       const float* __restrict__ w1, const float* __restrict__ b1,
                       const float* __restrict__ w2, const float* __restrict__ b2,
                       const float* __restrict__ g, const float* __restrict__ be,
                       float* __restrict__ y1)
{
    __shared__ float t1[CDY]; __shared__ float red[CDY];
    int b = blockIdx.x, t = threadIdx.x;
    float s = b1[t];
    for (int k = 0; k < CDY; ++k) s += new_y[b * CDY + k] * w1[k * CDY + t];
    t1[t] = fmaxf(s, 0.f);
    __syncthreads();
    float s2 = b2[t];
    for (int k = 0; k < CDY; ++k) s2 += t1[k] * w2[k * CDY + t];
    float v = y[b * CDY + t] + s2;
    red[t] = v; __syncthreads();
    for (int o = 32; o > 0; o >>= 1) { if (t < o) red[t] += red[t + o]; __syncthreads(); }
    float mean = red[0] * (1.f / 64.f); __syncthreads();
    float d = v - mean; red[t] = d * d; __syncthreads();
    for (int o = 32; o > 0; o >>= 1) { if (t < o) red[t] += red[t + o]; __syncthreads(); }
    float inv = rsqrtf(red[0] * (1.f / 64.f) + EPSLN);
    y1[b * CDY + t] = d * inv * g[t] + be[t];
}

__global__ void k_yffn1(const float* __restrict__ y1, const float* __restrict__ w,
                        const float* __restrict__ bb, float* __restrict__ HY)
{
    int b = blockIdx.x, t = threadIdx.x;
    for (int q = 0; q < CFFY / 256; ++q) {
        int n = q * 256 + t;
        float s = bb[n];
        for (int k = 0; k < CDY; ++k) s += y1[b * CDY + k] * w[k * CFFY + n];
        HY[b * CFFY + n] = fmaxf(s, 0.f);
    }
}

__global__ void k_yffn2(const float* __restrict__ y1, const float* __restrict__ HY,
                        const float* __restrict__ w, const float* __restrict__ bb,
                        const float* __restrict__ g, const float* __restrict__ be,
                        float* __restrict__ yf)
{
    __shared__ float red[CDY];
    int b = blockIdx.x, t = threadIdx.x;
    float s = bb[t];
    for (int k = 0; k < CFFY; ++k) s += HY[b * CFFY + k] * w[k * CDY + t];
    float v = y1[b * CDY + t] + s;
    red[t] = v; __syncthreads();
    for (int o = 32; o > 0; o >>= 1) { if (t < o) red[t] += red[t + o]; __syncthreads(); }
    float mean = red[0] * (1.f / 64.f); __syncthreads();
    float d = v - mean; red[t] = d * d; __syncthreads();
    for (int o = 32; o > 0; o >>= 1) { if (t < o) red[t] += red[t + o]; __syncthreads(); }
    float inv = rsqrtf(red[0] * (1.f / 64.f) + EPSLN);
    yf[b * CDY + t] = d * inv * g[t] + be[t];
}

// ------------------------------ pos path -----------------------------------

__global__ __launch_bounds__(32)
void k_msg(const float* __restrict__ msg1, const float* __restrict__ w2,
           float* __restrict__ msg)
{
    size_t e = blockIdx.x; int lane = threadIdx.x;
    float s = msg1[e * CDE + lane] * w2[lane] + msg1[e * CDE + 32 + lane] * w2[32 + lane];
#pragma unroll
    for (int o = 16; o > 0; o >>= 1) s += __shfl_xor(s, o, 32);
    if (lane == 0) msg[e] = s;
}

__global__ __launch_bounds__(32)
void k_vel(const float* __restrict__ msg, const float* __restrict__ posm,
           const unsigned char* __restrict__ mask, float* __restrict__ vel)
{
    int ni = blockIdx.x, b = ni >> 7, lane = threadIdx.x;
    float px = posm[ni * 3], py = posm[ni * 3 + 1], pz = posm[ni * 3 + 2];
    float sx = 0.f, sy = 0.f, sz = 0.f;
    for (int q = 0; q < 4; ++q) {
        int j = q * 32 + lane, nj = b * CN + j;
        float m = msg[(size_t)ni * CN + j];
        sx += m * (px - posm[nj * 3]); sy += m * (py - posm[nj * 3 + 1]); sz += m * (pz - posm[nj * 3 + 2]);
    }
#pragma unroll
    for (int o = 16; o > 0; o >>= 1) { sx += __shfl_xor(sx, o, 32); sy += __shfl_xor(sy, o, 32); sz += __shfl_xor(sz, o, 32); }
    if (lane == 0) {
        float m = mask[ni] ? 1.f : 0.f;
        vel[ni * 3] = sx * m; vel[ni * 3 + 1] = sy * m; vel[ni * 3 + 2] = sz * m;
    }
}

__global__ void k_pos_final(const float* __restrict__ vel, const float* __restrict__ posm,
                            const unsigned char* __restrict__ mask, const float* __restrict__ nval,
                            const float* __restrict__ se3, float* __restrict__ newpos)
{
    __shared__ float rx[CN], ry[CN], rz[CN];
    int b = blockIdx.x, i = threadIdx.x, ni = b * CN + i;
    float nv = nval[b];
    float vx = vel[ni * 3], vy = vel[ni * 3 + 1], vz = vel[ni * 3 + 2];
    rx[i] = vx; ry[i] = vy; rz[i] = vz; __syncthreads();
    for (int o = 64; o > 0; o >>= 1) { if (i < o) { rx[i] += rx[i + o]; ry[i] += ry[i + o]; rz[i] += rz[i + o]; } __syncthreads(); }
    float mvx = rx[0] / nv, mvy = ry[0] / nv, mvz = rz[0] / nv; __syncthreads();
    float m = mask[ni] ? 1.f : 0.f;
    vx = (vx - mvx) * m; vy = (vy - mvy) * m; vz = (vz - mvz) * m;
    float s2 = vx * vx + vy * vy + vz * vz;
    float vn = s2 > 0.f ? sqrtf(s2) : 0.f;
    rx[i] = vn; __syncthreads();
    for (int o = 64; o > 0; o >>= 1) { if (i < o) rx[i] += rx[i + o]; __syncthreads(); }
    float mean_norm = rx[0] / nv; __syncthreads();
    float w = se3[0] / (mean_norm + EPSLN);
    float nx = w * vx + posm[ni * 3], ny = w * vy + posm[ni * 3 + 1], nz = w * vz + posm[ni * 3 + 2];
    rx[i] = nx; ry[i] = ny; rz[i] = nz; __syncthreads();
    for (int o = 64; o > 0; o >>= 1) { if (i < o) { rx[i] += rx[i + o]; ry[i] += ry[i + o]; rz[i] += rz[i + o]; } __syncthreads(); }
    float ax = rx[0] / nv, ay = ry[0] / nv, az = rz[0] / nv;
    newpos[ni * 3] = (nx - ax) * m; newpos[ni * 3 + 1] = (ny - ay) * m; newpos[ni * 3 + 2] = (nz - az) * m;
}

// ---------------------------------------------------------------------------
// host orchestration
// ---------------------------------------------------------------------------
extern "C" void kernel_launch(void* const* d_in, const int* in_sizes, int n_in,
                              void* d_out, int out_size, void* d_ws, size_t ws_size,
                              hipStream_t stream)
{
    (void)in_sizes; (void)n_in; (void)out_size; (void)ws_size;
    const float* X   = (const float*)d_in[0];
    const float* E   = (const float*)d_in[1];
    const float* yIn = (const float*)d_in[2];
    const float* pos = (const float*)d_in[3];
    const unsigned char* nmask = (const unsigned char*)d_in[4];
#define P(i) ((const float*)d_in[i])
    const float *W_inE = P(5),  *B_inE = P(6);
    const float *W_xe1 = P(7),  *B_xe1 = P(8);
    const float *W_xe2 = P(9),  *B_xe2 = P(10);
    const float *W_d1  = P(11), *B_d1  = P(12);
    const float *W_np1 = P(13), *B_np1 = P(14);
    const float *W_np2 = P(15), *B_np2 = P(16);
    const float *W_da  = P(17), *B_da  = P(18);
    const float *W_dm  = P(19), *B_dm  = P(20);
    const float *W_k   = P(21), *B_k   = P(22);
    const float *W_q   = P(23), *B_q   = P(24);
    const float *W_v   = P(25), *B_v   = P(26);
    const float *W_a   = P(27);
    const float *W_out = P(28), *B_out = P(29);
    const float *W_eam = P(30), *B_eam = P(31);
    const float *W_pam = P(32), *B_pam = P(33);
    const float *W_exm = P(34), *B_exm = P(35);
    const float *W_pxm = P(36), *B_pxm = P(37);
    const float *W_yem = P(38), *B_yem = P(39);
    const float *W_yea = P(40), *B_yea = P(41);
    const float *W_yxm = P(42), *B_yxm = P(43);
    const float *W_yxa = P(44), *B_yxa = P(45);
    const float *W_yy  = P(46), *B_yy  = P(47);
    const float *W_xy  = P(48), *B_xy  = P(49);
    const float *W_ey  = P(50), *B_ey  = P(51);
    const float *W_dy  = P(52), *B_dy  = P(53);
    const float *W_ep1 = P(54);
    const float *W_ep2 = P(55);
    const float *W_xo  = P(56), *B_xo  = P(57);
    const float *W_eo  = P(58), *B_eo  = P(59);
    const float *W_yo1 = P(60), *B_yo1 = P(61);
    const float *W_yo2 = P(62), *B_yo2 = P(63);
    const float *W_lx1 = P(64), *B_lx1 = P(65);
    const float *W_lx2 = P(66), *B_lx2 = P(67);
    const float *W_le1 = P(68), *B_le1 = P(69);
    const float *W_le2 = P(70), *B_le2 = P(71);
    const float *W_ly1 = P(72), *B_ly1 = P(73);
    const float *W_ly2 = P(74), *B_ly2 = P(75);
    const float *gX1 = P(76), *bX1 = P(77);
    const float *gX2 = P(78), *bX2 = P(79);
    const float *gE1 = P(80), *bE1 = P(81);
    const float *gE2 = P(82), *bE2 = P(83);
    const float *gY1 = P(84), *bY1 = P(85);
    const float *gY2 = P(86), *bY2 = P(87);
    const float *se3 = P(88);
#undef P
    float* outXf  = (float*)d_out;
    float* outEf  = outXf + (size_t)NODES * CDX;
    float* outYf  = outEf + (size_t)EDGESI * CDE;
    float* outPos = outYf + (size_t)CB * CDY;

    float* Wsp = (float*)d_ws; size_t off = 0;
    auto alloc = [&](size_t n) { float* p = Wsp + off; off += n; return p; };
    float* Qb    = alloc((size_t)NODES * CDX);
    float* Kb    = alloc((size_t)NODES * CDX);
    float* Vb    = alloc((size_t)NODES * CDX);
    float* xe1   = alloc((size_t)NODES * CDE);
    float* xe2   = alloc((size_t)NODES * CDE);
    float* np1   = alloc((size_t)NODES * CDE);
    float* np2   = alloc((size_t)NODES * CDE);
    float* posm  = alloc((size_t)NODES * 3);
    float* npn   = alloc(NODES);
    float* nval  = alloc(CB);
    float* ye_mul = alloc(CB * CDE); float* ye_add = alloc(CB * CDE);
    float* yx_mul = alloc(CB * CDX); float* yx_add = alloc(CB * CDX);
    float* dist1 = alloc((size_t)EDGESI * CDE);
    float* Ybuf  = alloc((size_t)EDGESI * CDE);
    float* Econd = alloc((size_t)EDGESI * CDE);
    float* tmpA  = alloc((size_t)EDGESI * CDE);
    float* tmpB  = alloc((size_t)EDGESI * CDE);
    float* ea    = alloc((size_t)EDGESI * CNH);
    float* pa    = alloc((size_t)EDGESI * CNH);
    float* a0    = alloc((size_t)EDGESI * CNH);   // (b,h,i,j)
    float* alp   = alloc((size_t)EDGESI * CNH);   // (b,h,i,j)
    float* msg   = alloc(EDGESI);
    float* wV    = alloc((size_t)NODES * CNH * CDX);
    float* wVo   = alloc((size_t)NODES * CDX);
    float* cx1   = alloc((size_t)NODES * 256);
    float* exm   = alloc((size_t)NODES * CDX);
    float* cx2   = alloc((size_t)NODES * 256);
    float* pxm   = alloc((size_t)NODES * CDX);
    float* newX  = alloc((size_t)NODES * CDX);
    float* Xout  = alloc((size_t)NODES * CDX);
    float* X1    = alloc((size_t)NODES * CDX);
    float* HX    = alloc((size_t)NODES * CFFX);
    float* XF2   = alloc((size_t)NODES * CDX);
    float* st    = alloc((size_t)4 * NODES * CDE);
    float* eyY   = alloc(CB * 256);
    float* eyD   = alloc(CB * 256);
    float* xst   = alloc(CB * 1024);
    float* new_y = alloc(CB * CDY);
    float* y1    = alloc(CB * CDY);
    float* HY    = alloc(CB * CFFY);
    float* vel   = alloc((size_t)NODES * 3);
    unsigned char* em = (unsigned char*)alloc((EDGESI + 3) / 4 + 64);

    auto gemm = [&](const float* A, int lda, const float* Wt, int ldw, const float* bias,
                    float* C, int ldc, int M, int Nn, int K,
                    const unsigned char* rm, int relu) {
        dim3 g(Nn / 64, M / 64);
        k_gemm<<<g, 128, 0, stream>>>(A, lda, Wt, ldw, bias, C, ldc, M, Nn, K, rm, relu);
    };
    auto gemm_s = [&](const float* A, int lda, const float* Wt, int ldw, const float* bias,
                      float* C, int ldc, int M, int Nn, int K) {
        k_gemm_s<<<M / 16, 32, 0, stream>>>(A, lda, Wt, ldw, bias, C, ldc, M, Nn, K);
    };

    // ---- prep ----
    k_nval<<<CB, CN, 0, stream>>>(nmask, nval);
    k_edgemask<<<EDGESI / 256, 256, 0, stream>>>(nmask, em);
    k_prep_node<<<NODES, CDE, 0, stream>>>(pos, nmask, W_np1, B_np1, W_np2, B_np2, posm, npn, np1, np2);
    k_ylin<<<CB, CDX, 0, stream>>>(yIn, W_yem, B_yem, W_yea, B_yea, W_yxm, B_yxm, W_yxa, B_yxa,
                                   ye_mul, ye_add, yx_mul, yx_add);
    // ---- node linears (WMMA) ----
    gemm(X, CDX, W_xe1, CDE, B_xe1, xe1, CDE, NODES, CDE, CDX, nmask, 0);
    gemm(X, CDX, W_xe2, CDE, B_xe2, xe2, CDE, NODES, CDE, CDX, nmask, 0);
    gemm(X, CDX, W_q, CDX, B_q, Qb, CDX, NODES, CDX, CDX, nmask, 0);
    gemm(X, CDX, W_k, CDX, B_k, Kb, CDX, NODES, CDX, CDX, nmask, 0);
    gemm(X, CDX, W_v, CDX, B_v, Vb, CDX, NODES, CDX, CDX, nmask, 0);
    // ---- edge geometry & edge linears ----
    k_edge_geom<<<EDGESI, CDE, 0, stream>>>(posm, npn, np1, np2, nmask, W_d1, B_d1, dist1);
    gemm(E, CDE, W_inE, CDE, B_inE, Ybuf, CDE, EDGESI, CDE, CDE, nullptr, 0);
    gemm(dist1, CDE, W_da, CDE, B_da, tmpA, CDE, EDGESI, CDE, CDE, nullptr, 0);
    gemm(dist1, CDE, W_dm, CDE, B_dm, tmpB, CDE, EDGESI, CDE, CDE, nullptr, 0);
    k_edge_combine<<<(unsigned)((size_t)EDGESI * CDE / 256), 256, 0, stream>>>(
        Ybuf, tmpA, tmpB, xe1, xe2, ye_mul, ye_add, nmask, Ybuf, Econd);
    gemm(Ybuf, CDE, W_eo, CDE, B_eo, tmpA, CDE, EDGESI, CDE, CDE, em, 0);          // Eout
    gemm(Ybuf, CDE, W_ep1, CDE, nullptr, tmpB, CDE, EDGESI, CDE, CDE, nullptr, 1); // relu(e_pos1)
    k_msg<<<EDGESI, 32, 0, stream>>>(tmpB, W_ep2, msg);
    gemm_s(Econd, CDE, W_eam, CNH, B_eam, ea, CNH, EDGESI, CNH, CDE);
    gemm_s(dist1, CDE, W_pam, CNH, B_pam, pa, CNH, EDGESI, CNH, CDE);
    // ---- attention (WMMA) ----
    k_att_logits<<<EDGESI / 16, 32, 0, stream>>>(Qb, Kb, W_a, a0);
    k_att_softmax<<<NODES * CNH, 32, 0, stream>>>(a0, ea, pa, nmask, alp);
    k_wV<<<dim3(CDX / 16, CN / 16, CB * CNH), 32, 0, stream>>>(alp, Vb, wV);
    gemm(wV, CNH * CDX, W_out, CDX, B_out, wVo, CDX, NODES, CDX, CNH * CDX, nmask, 0);
    // ---- edge->node gains, newX, Xout ----
    k_etox<<<NODES, CDE, 0, stream>>>(Econd, nmask, nval, cx1);
    gemm(cx1, 256, W_exm, CDX, B_exm, exm, CDX, NODES, CDX, 256, nullptr, 0);
    k_etox<<<NODES, CDE, 0, stream>>>(dist1, nmask, nval, cx2);
    gemm(cx2, 256, W_pxm, CDX, B_pxm, pxm, CDX, NODES, CDX, 256, nullptr, 0);
    k_newX<<<NODES, CDX, 0, stream>>>(wVo, exm, pxm, yx_mul, yx_add, newX);
    gemm(newX, CDX, W_xo, CDX, B_xo, Xout, CDX, NODES, CDX, CDX, nmask, 0);
    // ---- X residual + FFN + LN ----
    k_ln_node<<<NODES, 32, 0, stream>>>(X, Xout, gX1, bX1, X1);
    gemm(X1, CDX, W_lx1, CFFX, B_lx1, HX, CFFX, NODES, CFFX, CDX, nullptr, 1);
    gemm(HX, CFFX, W_lx2, CDX, B_lx2, XF2, CDX, NODES, CDX, CFFX, nullptr, 0);
    k_ln_node<<<NODES, 32, 0, stream>>>(X1, XF2, gX2, bX2, outXf);
    // ---- y path ----
    k_etoy1<<<NODES, CDE, 0, stream>>>(Ybuf, nmask, st);
    k_etoy2<<<CB, CDE, 0, stream>>>(st, nval, eyY);
    k_etoy1<<<NODES, CDE, 0, stream>>>(dist1, nmask, st);
    k_etoy2<<<CB, CDE, 0, stream>>>(st, nval, eyD);
    k_xtoy<<<CB, CDX, 0, stream>>>(newX, nmask, nval, xst);
    k_ycombine<<<CB, CDY, 0, stream>>>(yIn, xst, eyY, eyD, W_yy, B_yy, W_xy, B_xy,
                                       W_ey, B_ey, W_dy, B_dy, new_y);
    k_yout<<<CB, CDY, 0, stream>>>(yIn, new_y, W_yo1, B_yo1, W_yo2, B_yo2, gY1, bY1, y1);
    k_yffn1<<<CB, 256, 0, stream>>>(y1, W_ly1, B_ly1, HY);
    k_yffn2<<<CB, CDY, 0, stream>>>(y1, HY, W_ly2, B_ly2, gY2, bY2, outYf);
    // ---- E residual + fused FFN + LN + symmetrize ----
    k_ln_edge<<<EDGESI, 32, 0, stream>>>(E, tmpA, gE1, bE1, tmpB);            // E1
    k_effn<<<EDGESI / 16, 32, 0, stream>>>(tmpB, W_le1, B_le1, W_le2, B_le2, tmpA);
    k_ln_edge<<<EDGESI, 32, 0, stream>>>(tmpB, tmpA, gE2, bE2, Ybuf);         // Ef (pre-sym)
    k_sym<<<EDGESI, CDE, 0, stream>>>(Ybuf, outEf);
    // ---- pos update ----
    k_vel<<<NODES, 32, 0, stream>>>(msg, posm, nmask, vel);
    k_pos_final<<<CB, CN, 0, stream>>>(vel, posm, nmask, nval, se3, outPos);
}
